// Codebook_46514495816391
// MI455X (gfx1250) — matrix operand order repaired and checked by reading, not compiled, not run on hardware
//
#include <hip/hip_runtime.h>
#include <hip/hip_bf16.h>

typedef _Float16 f16;
typedef __attribute__((ext_vector_type(8)))  _Float16 v8h;
typedef __attribute__((ext_vector_type(16))) _Float16 v16h;
typedef __attribute__((ext_vector_type(8)))  float    v8f;
typedef int v4i __attribute__((vector_size(16)));

#define N_TOK   32768
#define DIMC    256
#define SSIZE   1024
#define M_BLK   128      // tokens per block
#define LDA     40       // padded LDS row stride in halves (32 K + 8 pad)

// ---- CDNA5 async global->LDS path (guarded; falls back to sync copy) ----
#if __has_builtin(__builtin_amdgcn_global_load_async_to_lds_b128) && \
    __has_builtin(__builtin_amdgcn_s_wait_asynccnt)
#define USE_ASYNC_LDS 1
#else
#define USE_ASYNC_LDS 0
#endif

typedef __attribute__((address_space(1))) v4i* gv4i_p;   // global int4*
typedef __attribute__((address_space(3))) v4i* lv4i_p;   // LDS int4*

// ---------------- helpers ----------------

__device__ __forceinline__ v16h ld_fragA(const f16* p) {
  // elements 0..7 at p, elements 8..15 at p+16 (A 16x32 f16 layout)
  v8h a = *(const v8h*)p;
  v8h b = *(const v8h*)(p + 16);
  return __builtin_shufflevector(a, b, 0,1,2,3,4,5,6,7,8,9,10,11,12,13,14,15);
}

__device__ __forceinline__ v16h ld_fragB(const f16* p) {
  // 16 contiguous halves (per-lane contiguous K order; unused, B is loaded with ld_fragA)
  v8h a = *(const v8h*)p;
  v8h b = *(const v8h*)(p + 8);
  return __builtin_shufflevector(a, b, 0,1,2,3,4,5,6,7,8,9,10,11,12,13,14,15);
}

__device__ __forceinline__ v8f wmma_f16(v16h a, v16h b, v8f c) {
  return __builtin_amdgcn_wmma_f32_16x16x32_f16(false, a, false, b, (short)0, c, false, false);
}

// ---------------- prep: w2 + f16 hi/lo codebook + zero hist ----------------

__global__ __launch_bounds__(64) void vq_prep(const float* __restrict__ w,
                                              float* __restrict__ w2,
                                              unsigned int* __restrict__ hist,
                                              f16* __restrict__ wh,
                                              f16* __restrict__ wl) {
  const int s = blockIdx.x;          // 0..1023
  const int t = threadIdx.x;         // 0..63
  float p = 0.0f;
  for (int c = t; c < DIMC; c += 64) {
    float v = w[(size_t)s * DIMC + c];
    f16 h = (f16)v;
    wh[(size_t)s * DIMC + c] = h;
    wl[(size_t)s * DIMC + c] = (f16)(v - (float)h);
    p += v * v;
  }
  __shared__ float r[64];
  r[t] = p;
  __syncthreads();
  for (int q = 32; q > 0; q >>= 1) {
    if (t < q) r[t] += r[t + q];
    __syncthreads();
  }
  if (t == 0) {
    w2[s] = r[0];
    hist[s] = 0u;
  }
}

// ---------------- main: split-f16 WMMA GEMM + argmin + gather ----------------

__global__ __launch_bounds__(256) void vq_main(const float* __restrict__ x,
                                               const float* __restrict__ w,
                                               const f16* __restrict__ wh,
                                               const f16* __restrict__ wl,
                                               const float* __restrict__ w2g,
                                               unsigned int* __restrict__ hist,
                                               float* __restrict__ partial,
                                               float* __restrict__ out) {
  __shared__ float w2s[SSIZE];
  __shared__ __align__(32) unsigned char smem_raw[40960];

  f16* Ah = (f16*)(smem_raw);            // 128 x 40 halves = 10240 B
  f16* Al = (f16*)(smem_raw + 10240);
  f16* Bh = (f16*)(smem_raw + 20480);
  f16* Bl = (f16*)(smem_raw + 30720);

  const int tid   = threadIdx.x;
  const int lane  = tid & 31;
  const int wv    = tid >> 5;           // 0..7
  const int waveM = wv >> 1;            // 0..3  -> M-tiles {2wM, 2wM+1}
  const int waveN = wv & 1;             // 0..1  -> N-tiles {4wN .. 4wN+3}
  const int lhalf = lane >> 4;          // 0/1
  const int l15   = lane & 15;

  const int n0  = blockIdx.x * M_BLK;   // first token of block (one image b)
  const int bb  = n0 >> 10;
  const int hw0 = n0 & 1023;
  const size_t xbase = (size_t)bb * DIMC * 1024 + hw0;

  for (int i = tid; i < SSIZE; i += 256) w2s[i] = w2g[i];
  __syncthreads();

  float best[2][8];
  int   bcode[2][8];
#pragma unroll
  for (int t = 0; t < 2; ++t)
#pragma unroll
    for (int i = 0; i < 8; ++i) { best[t][i] = 3.4e38f; bcode[t][i] = 0; }

  const v8f vzero = {0.f,0.f,0.f,0.f,0.f,0.f,0.f,0.f};

  for (int cc = 0; cc < 8; ++cc) {            // 8 code-chunks of 128 codes
    v8f acc[2][4];
#pragma unroll
    for (int t = 0; t < 2; ++t)
#pragma unroll
      for (int u = 0; u < 4; ++u) acc[t][u] = vzero;

    for (int kc = 0; kc < 8; ++kc) {          // K-chunks of 32
      __syncthreads();                        // previous chunk fully consumed
      // ---- stage A: x[128 tok][32 k] -> f16 hi/lo (coalesced over tokens)
      {
        const int token = tid & 127;
        const int ch0   = tid >> 7;           // 0/1
#pragma unroll
        for (int cl = ch0; cl < 32; cl += 2) {
          float v = x[xbase + (size_t)(kc * 32 + cl) * 1024 + token];
          f16 h = (f16)v;
          f16 l = (f16)(v - (float)h);
          Ah[token * LDA + cl] = h;
          Al[token * LDA + cl] = l;
        }
      }
      // ---- stage B: async DMA pre-converted f16 hi/lo tiles into LDS
      {
        const int code_l = tid >> 1;          // 0..127
        const int q0     = (tid & 1) * 2;     // chunk-of-8-halves index base
#pragma unroll
        for (int j = 0; j < 2; ++j) {
          const int q = q0 + j;               // 0..3
          const size_t goff = (size_t)(cc * 128 + code_l) * DIMC + kc * 32 + q * 8;
          f16* lh = Bh + code_l * LDA + q * 8;
          f16* ll = Bl + code_l * LDA + q * 8;
#if USE_ASYNC_LDS
          __builtin_amdgcn_global_load_async_to_lds_b128((gv4i_p)(wh + goff), (lv4i_p)lh, 0, 0);
          __builtin_amdgcn_global_load_async_to_lds_b128((gv4i_p)(wl + goff), (lv4i_p)ll, 0, 0);
#else
          *(v8h*)lh = *(const v8h*)(wh + goff);
          *(v8h*)ll = *(const v8h*)(wl + goff);
#endif
        }
      }
#if USE_ASYNC_LDS
      __builtin_amdgcn_s_wait_asynccnt(0);
#endif
      __syncthreads();

      // ---- fragments + 24 WMMA (hi*hi + lo*hi + hi*lo)
      v16h a_h[2], a_l[2], b_h[4], b_l[4];
#pragma unroll
      for (int t = 0; t < 2; ++t) {
        const int row = (2 * waveM + t) * 16 + l15;
        a_h[t] = ld_fragA(Ah + row * LDA + lhalf * 8);
        a_l[t] = ld_fragA(Al + row * LDA + lhalf * 8);
      }
#pragma unroll
      for (int u = 0; u < 4; ++u) {
        const int crow = (waveN * 4 + u) * 16 + l15;
        b_h[u] = ld_fragA(Bh + crow * LDA + lhalf * 8);
        b_l[u] = ld_fragA(Bl + crow * LDA + lhalf * 8);
      }
#pragma unroll
      for (int t = 0; t < 2; ++t)
#pragma unroll
        for (int u = 0; u < 4; ++u) {
          acc[t][u] = wmma_f16(a_h[t], b_h[u], acc[t][u]);
          acc[t][u] = wmma_f16(a_l[t], b_h[u], acc[t][u]);
          acc[t][u] = wmma_f16(a_h[t], b_l[u], acc[t][u]);
        }
    }

    // ---- per-lane argmin update for this code-chunk
#pragma unroll
    for (int t = 0; t < 2; ++t)
#pragma unroll
      for (int u = 0; u < 4; ++u) {
        const int code = cc * 128 + (waveN * 4 + u) * 16 + l15;
        const float d2base = w2s[code];
#pragma unroll
        for (int i = 0; i < 8; ++i) {
          float d2 = d2base - 2.0f * acc[t][u][i];
          if (d2 < best[t][i] || (d2 == best[t][i] && code < bcode[t][i])) {
            best[t][i]  = d2;
            bcode[t][i] = code;
          }
        }
      }
  }

  __syncthreads();   // GEMM staging done for all waves; reuse LDS as scoreboard

  float* sv = (float*)smem_raw;                  // [128][32]
  int*   si = (int*)(smem_raw + 16384);          // [128][32]
#pragma unroll
  for (int t = 0; t < 2; ++t)
#pragma unroll
    for (int i = 0; i < 8; ++i) {
      const int rowl = (2 * waveM + t) * 16 + i + lhalf * 8;
      const int slot = waveN * 16 + l15;
      sv[rowl * 32 + slot] = best[t][i];
      si[rowl * 32 + slot] = bcode[t][i];
    }
  __syncthreads();

  int* idx_row = (int*)(smem_raw + 32768);       // [128]
  if (tid < 128) {
    float bv = sv[tid * 32];
    int   bi = si[tid * 32];
    for (int s = 1; s < 32; ++s) {
      float v = sv[tid * 32 + s];
      int   d = si[tid * 32 + s];
      if (v < bv || (v == bv && d < bi)) { bv = v; bi = d; }
    }
    idx_row[tid] = bi;
    atomicAdd(&hist[bi], 1u);
  }
  __syncthreads();

  // ---- output gather + commitment-loss partial
  float lsum = 0.0f;
  const int token4 = (tid & 31) * 4;             // 0..124
  const int c0     = tid >> 5;                   // 0..7
  const int r0 = idx_row[token4 + 0];
  const int r1 = idx_row[token4 + 1];
  const int r2 = idx_row[token4 + 2];
  const int r3 = idx_row[token4 + 3];
  for (int c = c0; c < DIMC; c += 8) {
    const size_t off = xbase + (size_t)c * 1024 + token4;
    const float4 xv = *(const float4*)(x + off);
    float4 q;
    q.x = w[(size_t)r0 * DIMC + c];
    q.y = w[(size_t)r1 * DIMC + c];
    q.z = w[(size_t)r2 * DIMC + c];
    q.w = w[(size_t)r3 * DIMC + c];
    *(float4*)(out + off) = q;
    float dx = q.x - xv.x, dy = q.y - xv.y, dz = q.z - xv.z, dw = q.w - xv.w;
    lsum += dx * dx + dy * dy + dz * dz + dw * dw;
  }

  float* red = (float*)(smem_raw + 33280);       // [256]
  red[tid] = lsum;
  __syncthreads();
  for (int s = 128; s > 0; s >>= 1) {
    if (tid < s) red[tid] += red[tid + s];
    __syncthreads();
  }
  if (tid == 0) partial[blockIdx.x] = red[0];
}

// ---------------- finalize: perplexity + loss scalars ----------------

__global__ __launch_bounds__(1024) void vq_fin(const unsigned int* __restrict__ hist,
                                               const float* __restrict__ partial,
                                               float* __restrict__ outs) {
  __shared__ float r[1024];
  const int t = threadIdx.x;
  float p = (float)hist[t] * (1.0f / 32768.0f);
  r[t] = p * logf(p + 1e-6f);
  __syncthreads();
  for (int s = 512; s > 0; s >>= 1) {
    if (t < s) r[t] += r[t + s];
    __syncthreads();
  }
  float ent = 0.0f;
  if (t == 0) ent = r[0];
  __syncthreads();
  r[t] = (t < 256) ? partial[t] : 0.0f;
  __syncthreads();
  for (int s = 512; s > 0; s >>= 1) {
    if (t < s) r[t] += r[t + s];
    __syncthreads();
  }
  if (t == 0) {
    outs[0] = 0.25f * r[0] / (32768.0f * 256.0f);  // BETA * mean mse
    outs[1] = expf(-ent);                           // perplexity
  }
}

// ---------------- launch ----------------

extern "C" void kernel_launch(void* const* d_in, const int* in_sizes, int n_in,
                              void* d_out, int out_size, void* d_ws, size_t ws_size,
                              hipStream_t stream) {
  (void)in_sizes; (void)n_in; (void)out_size; (void)ws_size;
  const float* x = (const float*)d_in[0];     // [32,256,32,32] f32
  const float* w = (const float*)d_in[1];     // [1024,256] f32

  float*        w2      = (float*)d_ws;                              // 1024 f32
  unsigned int* hist    = (unsigned int*)((char*)d_ws + 4096);       // 1024 u32
  float*        partial = (float*)((char*)d_ws + 8192);              // 256 f32
  f16*          wh      = (f16*)((char*)d_ws + 16384);               // 1024x256 f16
  f16*          wl      = (f16*)((char*)d_ws + 16384 + 524288);      // 1024x256 f16

  float* out = (float*)d_out;                 // [32,256,32,32] + 2 scalars

  vq_prep<<<SSIZE, 64, 0, stream>>>(w, w2, hist, wh, wl);
  vq_main<<<N_TOK / M_BLK, 256, 0, stream>>>(x, w, wh, wl, w2, hist, partial, out);
  vq_fin<<<1, 1024, 0, stream>>>(hist, partial, out + (size_t)N_TOK * DIMC);
}
